// spixel_conv2_5d_78752520339848
// MI455X (gfx1250) — matrix operand order repaired and checked by reading, not compiled, not run on hardware
//
#include <hip/hip_runtime.h>
#include <hip/hip_bf16.h>

// ---------------------------------------------------------------------------
// spixel_conv2_5d for MI455X (gfx1250, wave32, WMMA bf16)
//
//  k_tot_uv   : per-4x4-cell prob sum (tot) + uv output            (tiny)
//  k_wprep<>  : f32 weights (OIDHW) -> bf16 B-fragment layout      (tiny)
//  k_feat     : fused 9-neighbor gather * prob / tot -> bf16
//               channel-last feature volume [D*H*W][96]            (151 MB)
//  k_conv<>   : implicit-GEMM conv, v_wmma_f32_16x16x32_bf16,
//               per-wave 16x16 M/N tiles, NT accumulators,
//               branch-free (clamp+mask) boundary handling
// ---------------------------------------------------------------------------

typedef __bf16 v16bf __attribute__((ext_vector_type(16)));
typedef float  v8f   __attribute__((ext_vector_type(8)));
typedef unsigned int v8u __attribute__((ext_vector_type(8)));

__device__ __forceinline__ unsigned short f2bf(float f) {
  unsigned int u = __builtin_bit_cast(unsigned int, f);
  u += 0x7fffu + ((u >> 16) & 1u);          // round-to-nearest-even
  return (unsigned short)(u >> 16);
}

// ------------------------- constants (problem sizes) -----------------------
#define DD 24
#define HH 128
#define WW 256
#define HS 32          // HH/4
#define WS 64          // WW/4
#define M0 (DD*HH*WW)  // 786432 stem spatial
#define M1 (12*64*128) // 98304
#define M2 (6*32*64)   // 12288

// ------------------------- tot + uv kernel ---------------------------------
__global__ void k_tot_uv(const float* __restrict__ sp,
                         const float* __restrict__ xy,
                         float* __restrict__ tot,
                         float* __restrict__ uv_out) {
  int cell = blockIdx.x * blockDim.x + threadIdx.x;
  if (cell >= HS * WS) return;
  int by = cell / WS, bx = cell % WS;
  float ssum = 0.f, uxs = 0.f, uys = 0.f;
  for (int py = 0; py < 4; ++py) {
    for (int px = 0; px < 4; ++px) {
      int y = by * 4 + py, x = bx * 4 + px;
      float ps = 0.f, ux = 0.f, uy = 0.f;
      #pragma unroll
      for (int k = 0; k < 9; ++k) {
        int oy = 4 * ((k / 3) - 1), ox = 4 * ((k % 3) - 1);
        int sy = y - oy, sx = x - ox;
        if ((unsigned)sy < (unsigned)HH && (unsigned)sx < (unsigned)WW) {
          float p = sp[k * HH * WW + sy * WW + sx];
          ps += p;
          ux = xy[0 * HH * WW + sy * WW + sx] * p;   // last-write-wins
          uy = xy[1 * HH * WW + sy * WW + sx] * p;
        }
      }
      ssum += ps; uxs += ux; uys += uy;
    }
  }
  float t = ssum * (1.f / 16.f);
  tot[cell] = t;
  float inv = 1.f / t;
  uv_out[cell]           = uxs * (1.f / 16.f) * inv;
  uv_out[HS * WS + cell] = uys * (1.f / 16.f) * inv;
}

// ------------------------- weight swizzle ----------------------------------
// wb[kblk][n][j] bf16, K = kblk*16 + j = tap*CIN + ci  (tap-major, ci-minor)
template<int CIN, int COUT, int KD, int KH, int KW>
__global__ void k_wprep(const float* __restrict__ w,
                        unsigned short* __restrict__ wb, int total) {
  constexpr int KTOT = CIN * KD * KH * KW;
  int t = blockIdx.x * blockDim.x + threadIdx.x;
  if (t >= total) return;
  int j = t & 15;
  int n = (t >> 4) % COUT;
  int kblk = t / (16 * COUT);
  int K = kblk * 16 + j;
  float val = 0.f;
  if (K < KTOT) {
    int tap = K / CIN, ci = K % CIN;
    int kd = tap / (KH * KW), kh = (tap / KW) % KH, kw = tap % KW;
    val = w[((((size_t)n * CIN + ci) * KD + kd) * KH + kh) * KW + kw];
  }
  wb[t] = f2bf(val);
}

// ------------------------- fused gather/normalize --------------------------
// feat[(d*HH+y)*WW+x][96] bf16 ; channel = c*9 + k ; 72..95 zero padded
__global__ __launch_bounds__(256) void k_feat(const float* __restrict__ cv,
                                              const float* __restrict__ sp,
                                              const float* __restrict__ tot,
                                              unsigned short* __restrict__ feat) {
  int pos = blockIdx.x * blockDim.x + threadIdx.x;
  if (pos >= M0) return;
  int d = pos / (HH * WW);
  int r = pos % (HH * WW);
  int y = r / WW, x = r % WW;
  float inv = 1.f / tot[(y >> 2) * WS + (x >> 2)];
  unsigned int rowu[48];
  #pragma unroll
  for (int i = 0; i < 48; ++i) rowu[i] = 0u;
  #pragma unroll
  for (int k = 0; k < 9; ++k) {
    const int oy = 4 * ((k / 3) - 1), ox = 4 * ((k % 3) - 1);
    const int sy = y - oy, sx = x - ox;
    const bool ok = ((unsigned)sy < (unsigned)HH) && ((unsigned)sx < (unsigned)WW);
    const int sidx = ok ? (sy * WW + sx) : 0;        // clamp, mask via p=0
    const float p = ok ? sp[k * HH * WW + sidx] * inv : 0.f;
    #pragma unroll
    for (int c = 0; c < 8; ++c) {
      float v = cv[((size_t)c * DD + d) * (HH * WW) + sidx] * p;
      const int idx = c * 9 + k;
      rowu[idx >> 1] |= (unsigned)f2bf(v) << (16 * (idx & 1));
    }
  }
  uint4* dst = reinterpret_cast<uint4*>(feat + (size_t)pos * 96);
  #pragma unroll
  for (int i = 0; i < 12; ++i)
    dst[i] = make_uint4(rowu[4*i], rowu[4*i+1], rowu[4*i+2], rowu[4*i+3]);
}

// ------------------------- implicit-GEMM WMMA conv -------------------------
// act: [ID*IH*IW][CINPAD] bf16 bits ; wb: [2*KSTEPS][COUT][16] bf16 bits
// out_f32: [COUT][OD*OH*OW] ; act_next: [OD*OH*OW][CNEXT] bf16 bits
template<int CIN, int CINPAD, int COUT,
         int KD, int KH, int KW,
         int SD, int SH, int SW, int PD,
         int ID, int IH, int IW,
         int OD, int OH, int OW,
         bool WF32, bool WBF16, int CNEXT>
__global__ __launch_bounds__(256) void k_conv(
    const unsigned short* __restrict__ act,
    const unsigned short* __restrict__ wb,
    float* __restrict__ out_f32,
    unsigned short* __restrict__ act_next) {
  constexpr int KTOT   = KD * KH * KW * CIN;
  constexpr int KSTEPS = (KTOT + 31) / 32;
  constexpr int NT     = COUT / 16;
  constexpr int MTOT   = OD * OH * OW;
  constexpr int MT     = MTOT / 16;

  const int lane  = threadIdx.x & 31;
  const int wave  = threadIdx.x >> 5;
  const int mtile = blockIdx.x * 8 + wave;     // wave-uniform
  if (mtile >= MT) return;                     // whole-wave exit: EXEC stays all-1s

  const int lh   = lane >> 4;                  // lane half
  const int lidx = lane & 15;

  // A-row spatial decode (M = lidx for the A fragment)
  const int m  = mtile * 16 + lidx;
  const int od = m / (OH * OW);
  const int mr = m % (OH * OW);
  const int oh = mr / OW;
  const int ow = mr % OW;

  v8f acc[NT];
  #pragma unroll
  for (int nt = 0; nt < NT; ++nt)
    #pragma unroll
    for (int i = 0; i < 8; ++i) acc[nt][i] = 0.f;

  #pragma unroll
  for (int kb = 0; kb < KSTEPS; ++kb) {
    // ---- A fragment: 2 x 16B chunks per lane (K = kb*32 + 16*lh + j, matches B)
    // Branch-free: clamp address to 0 when out of tensor/pad range, then mask.
    uint4 c01[2];
    #pragma unroll
    for (int ch = 0; ch < 2; ++ch) {
      const int K0    = kb * 32 + lh * 16 + ch * 8;   // 8-contiguous K chunk
      const bool kok  = (K0 < KTOT);
      const int  tap  = kok ? (K0 / CIN) : 0;          // CIN%8==0 -> no straddle
      const int  ci   = kok ? (K0 % CIN) : 0;
      const int  kd   = tap / (KH * KW);
      const int  kh   = (tap / KW) % KH;
      const int  kw   = tap % KW;
      const int  id   = od * SD - PD + kd;
      const int  iy   = oh * SH + kh;
      const int  ix   = ow * SW + kw;
      const bool ok   = kok && ((unsigned)id < (unsigned)ID) &&
                        ((unsigned)iy < (unsigned)IH) &&
                        ((unsigned)ix < (unsigned)IW);
      const size_t off = ok ? ((size_t)(id * IH + iy) * IW + ix) * CINPAD + ci
                            : (size_t)0;               // clamped, always valid
      uint4 v = *reinterpret_cast<const uint4*>(act + off);  // global_load_b128
      const unsigned msk = ok ? 0xFFFFFFFFu : 0u;      // v_cndmask, no branches
      v.x &= msk; v.y &= msk; v.z &= msk; v.w &= msk;
      c01[ch] = v;
    }
    v8u au = { c01[0].x, c01[0].y, c01[0].z, c01[0].w,
               c01[1].x, c01[1].y, c01[1].z, c01[1].w };
    const v16bf afrag = __builtin_bit_cast(v16bf, au);

    if (kb + 1 < KSTEPS)                               // global_prefetch next B blk
      __builtin_prefetch(wb + ((size_t)((kb + 1) * 2 + lh) * COUT) * 16, 0, 3);

    #pragma unroll
    for (int nt = 0; nt < NT; ++nt) {
      // ---- B fragment: K = kb*32 + 16*lh + j, col = lidx; 32B contiguous
      const v8u* bp = reinterpret_cast<const v8u*>(
          wb + ((size_t)(kb * 2 + lh) * COUT + nt * 16 + lidx) * 16);
      const v16bf bfrag = __builtin_bit_cast(v16bf, *bp);
      acc[nt] = __builtin_amdgcn_wmma_f32_16x16x32_bf16(
          false, afrag, false, bfrag, (short)0, acc[nt], false, false);
    }
  }

  // ---- store: C/D layout row = 8*lh + r, col = lidx
  #pragma unroll
  for (int nt = 0; nt < NT; ++nt) {
    const int nn = nt * 16 + lidx;
    #pragma unroll
    for (int r = 0; r < 8; ++r) {
      float v = acc[nt][r];
      v = v > 0.f ? v : 0.f;                           // ReLU (BN identity at init)
      const int mm = mtile * 16 + 8 * lh + r;
      if constexpr (WF32)  out_f32[(size_t)nn * MTOT + mm] = v;
      if constexpr (WBF16) act_next[(size_t)mm * CNEXT + nn] = f2bf(v);
    }
  }
}

// ---------------------------------------------------------------------------
extern "C" void kernel_launch(void* const* d_in, const int* in_sizes, int n_in,
                              void* d_out, int out_size, void* d_ws, size_t ws_size,
                              hipStream_t stream) {
  (void)in_sizes; (void)n_in; (void)out_size; (void)ws_size;
  const float* cv    = (const float*)d_in[0];
  const float* sp    = (const float*)d_in[1];
  const float* xy    = (const float*)d_in[2];
  const float* wstem = (const float*)d_in[3];
  const float* w1a   = (const float*)d_in[4];
  const float* w1b   = (const float*)d_in[5];
  const float* w2a   = (const float*)d_in[6];
  const float* w2b   = (const float*)d_in[7];

  float* f1_out = (float*)d_out;                         // (1,32,12,64,128)
  float* f2_out = f1_out + (size_t)32 * M1;              // (1,64,6,32,64)
  float* uv_out = f2_out + (size_t)64 * M2;              // (1,2,2048)

  // workspace carve-up (256B aligned)
  char* ws = (char*)d_ws;
  size_t off = 0;
  auto take = [&](size_t bytes) -> char* {
    char* p = ws + off;
    off = (off + bytes + 255) & ~(size_t)255;
    return p;
  };
  float*          tot     = (float*)take((size_t)HS * WS * 4);
  unsigned short* wb_stem = (unsigned short*)take((size_t)6  * 16 * 16 * 2);
  unsigned short* wb_1a   = (unsigned short*)take((size_t)12 * 32 * 16 * 2);
  unsigned short* wb_1b   = (unsigned short*)take((size_t)6  * 32 * 16 * 2);
  unsigned short* wb_2a   = (unsigned short*)take((size_t)24 * 64 * 16 * 2);
  unsigned short* wb_2b   = (unsigned short*)take((size_t)12 * 64 * 16 * 2);
  unsigned short* feat    = (unsigned short*)take((size_t)M0 * 96 * 2);
  unsigned short* a_stem  = (unsigned short*)take((size_t)M0 * 16 * 2);
  unsigned short* a_1a    = (unsigned short*)take((size_t)M1 * 32 * 2);
  unsigned short* a_1b    = (unsigned short*)take((size_t)M1 * 32 * 2);
  unsigned short* a_2a    = (unsigned short*)take((size_t)M2 * 64 * 2);

  // tot + uv
  k_tot_uv<<<(HS * WS + 255) / 256, 256, 0, stream>>>(sp, xy, tot, uv_out);

  // weight swizzles
  k_wprep<72, 16, 1, 1, 1><<<(6  * 16 * 16 + 255) / 256, 256, 0, stream>>>(wstem, wb_stem, 6  * 16 * 16);
  k_wprep<16, 32, 3, 2, 2><<<(12 * 32 * 16 + 255) / 256, 256, 0, stream>>>(w1a,   wb_1a,   12 * 32 * 16);
  k_wprep<32, 32, 3, 1, 1><<<(6  * 32 * 16 + 255) / 256, 256, 0, stream>>>(w1b,   wb_1b,   6  * 32 * 16);
  k_wprep<32, 64, 3, 2, 2><<<(24 * 64 * 16 + 255) / 256, 256, 0, stream>>>(w2a,   wb_2a,   24 * 64 * 16);
  k_wprep<64, 64, 3, 1, 1><<<(12 * 64 * 16 + 255) / 256, 256, 0, stream>>>(w2b,   wb_2b,   12 * 64 * 16);

  // fused gather/normalize -> bf16 channel-last feature volume
  k_feat<<<(M0 + 255) / 256, 256, 0, stream>>>(cv, sp, tot, feat);

  // stem: 1x1x1, 72->16 (K padded to 96)
  k_conv<72, 96, 16, 1,1,1, 1,1,1, 0, 24,128,256, 24,128,256, false, true, 16>
      <<<(M0/16 + 7) / 8, 256, 0, stream>>>(feat, wb_stem, nullptr, a_stem);
  // w1a: (3,2,2)/s2 pad(1,0,0), 16->32
  k_conv<16, 16, 32, 3,2,2, 2,2,2, 1, 24,128,256, 12,64,128, false, true, 32>
      <<<(M1/16 + 7) / 8, 256, 0, stream>>>(a_stem, wb_1a, nullptr, a_1a);
  // w1b: (3,1,1) pad(1,0,0), 32->32 -> f1 (f32) + bf16 for next stage
  k_conv<32, 32, 32, 3,1,1, 1,1,1, 1, 12,64,128, 12,64,128, true, true, 32>
      <<<(M1/16 + 7) / 8, 256, 0, stream>>>(a_1a, wb_1b, f1_out, a_1b);
  // w2a: (3,2,2)/s2 pad(1,0,0), 32->64
  k_conv<32, 32, 64, 3,2,2, 2,2,2, 1, 12,64,128, 6,32,64, false, true, 64>
      <<<(M2/16 + 7) / 8, 256, 0, stream>>>(a_1b, wb_2a, nullptr, a_2a);
  // w2b: (3,1,1) pad(1,0,0), 64->64 -> f2 (f32)
  k_conv<64, 64, 64, 3,1,1, 1,1,1, 1, 6,32,64, 6,32,64, true, false, 16>
      <<<(M2/16 + 7) / 8, 256, 0, stream>>>(a_2a, wb_2b, f2_out, nullptr);
}
